// Mamba2Block_41721312313857
// MI455X (gfx1250) — matrix operand order repaired and checked by reading, not compiled, not run on hardware
//
#include <hip/hip_runtime.h>
#include <cstdint>
#include <cstddef>

#define TOK    4096      // B_SZ * SEQ
#define BSZ    2
#define LSEQ   2048
#define DM     128       // D_MODEL
#define DI     1024      // D_INNER
#define NH     8
#define HD     128
#define DSTATE 128
#define CONVD  1280      // D_INNER + 2*D_STATE
#define DIP    2312      // D_IN_PROJ
#define DIPP   2320      // padded to 145 * 16
#define NCH    16        // chunks per sequence (LSEQ / 128)
#define NBATCH 256       // BSZ * NH * NCH
#define QC     128       // chunk length

typedef unsigned short u16;
typedef __attribute__((ext_vector_type(8)))  unsigned short u16x8;
typedef __attribute__((ext_vector_type(16))) __bf16 v16bf;
typedef __attribute__((ext_vector_type(8)))  float v8f;

union Frag16 { v16bf v; u16x8 h[2]; };

__device__ __forceinline__ u16 f2bf(float f) {
  unsigned int u = __float_as_uint(f);
  unsigned int r = u + 0x7FFFu + ((u >> 16) & 1u);   // round-to-nearest-even
  return (u16)(r >> 16);
}

// ---------------------------------------------------------------------------
// f32 -> bf16 convert
// ---------------------------------------------------------------------------
__global__ void f32_to_bf16_kernel(const float* __restrict__ src,
                                   u16* __restrict__ dst, int n) {
  int i = blockIdx.x * blockDim.x + threadIdx.x;
  if (i < n) dst[i] = f2bf(src[i]);
}

// ---------------------------------------------------------------------------
// LayerNorm over 128 features -> bf16 (one block of 128 threads per token)
// ---------------------------------------------------------------------------
__global__ void __launch_bounds__(128)
ln_bf16_kernel(const float* __restrict__ x, const float* __restrict__ w,
               const float* __restrict__ b, u16* __restrict__ out) {
  int tok = blockIdx.x, tid = threadIdx.x;
  __shared__ float red[128];
  float v = x[(size_t)tok * DM + tid];
  red[tid] = v; __syncthreads();
  #pragma unroll
  for (int s = 64; s > 0; s >>= 1) { if (tid < s) red[tid] += red[tid + s]; __syncthreads(); }
  float mean = red[0] * (1.f / DM);
  __syncthreads();
  float xc = v - mean;
  red[tid] = xc * xc; __syncthreads();
  #pragma unroll
  for (int s = 64; s > 0; s >>= 1) { if (tid < s) red[tid] += red[tid + s]; __syncthreads(); }
  float var = red[0] * (1.f / DM);
  float y = xc * rsqrtf(var + 1e-5f) * w[tid] + b[tid];
  out[(size_t)tok * DM + tid] = f2bf(y);
}

// ---------------------------------------------------------------------------
// bf16 WMMA GEMM: C[M,N] = A[M,K] * W[N,K]^T   (+ residual or + bias)
// Wave computes a 16x64 strip. Padded N handled by clamping the weight-row
// pointer so the K-loop is guard-free; the store guard drops padded columns.
// mode 0: plain store  mode 1: += res[row*ldc+col]  mode 2: += bias[col]
// ---------------------------------------------------------------------------
__global__ void __launch_bounds__(128)
gemm_bf16_wmma(const u16* __restrict__ A, const u16* __restrict__ W,
               float* __restrict__ C, const float* __restrict__ res,
               const float* __restrict__ bias,
               int M, int N, int K, int ldc, int mode, int ngroups) {
  int gwave = (int)(((size_t)blockIdx.x * blockDim.x + threadIdx.x) >> 5);
  int lane  = threadIdx.x & 31;
  int mtiles = M >> 4;
  int mt = gwave % mtiles;
  int ng = gwave / mtiles;
  if (ng >= ngroups) return;              // wave-uniform exit
  int r0 = mt << 4;
  int n0 = ng * 64;
  int mrow = lane & 15;
  int hl   = lane >> 4;

  v8f acc[4];
  #pragma unroll
  for (int t = 0; t < 4; ++t) acc[t] = (v8f)(0.0f);

  const u16* arow = A + (size_t)(r0 + mrow) * K;
  int ncol[4]; const u16* wrow[4];
  #pragma unroll
  for (int t = 0; t < 4; ++t) {
    ncol[t] = n0 + t * 16 + mrow;
    int ncl = ncol[t] < N ? ncol[t] : (N - 1);   // clamp: guard-free loads
    wrow[t] = W + (size_t)ncl * K;
  }

  for (int k0 = 0; k0 < K; k0 += 32) {
    if (k0 + 32 < K) __builtin_prefetch(arow + k0 + 32, 0, 3);
    Frag16 af;                                   // A 16x32 bf16 layout
    af.h[0] = *(const u16x8*)(arow + k0 + hl * 8);
    af.h[1] = *(const u16x8*)(arow + k0 + 16 + hl * 8);
    #pragma unroll
    for (int t = 0; t < 4; ++t) {
      const u16* bp = wrow[t] + k0 + hl * 8;     // K 8h..8h+7 and 16+8h..16+8h+7 per lane-half
      Frag16 bf;
      bf.h[0] = *(const u16x8*)(bp);
      bf.h[1] = *(const u16x8*)(bp + 16);
      acc[t] = __builtin_amdgcn_wmma_f32_16x16x32_bf16(
          false, af.v, false, bf.v, (short)0, acc[t], false, false);
    }
  }

  #pragma unroll
  for (int t = 0; t < 4; ++t) {
    int col = ncol[t];
    if (col >= N) continue;
    #pragma unroll
    for (int i = 0; i < 8; ++i) {
      int row = r0 + i + (hl << 3);
      float v = acc[t][i];
      if (mode == 1)      v += res[(size_t)row * ldc + col];
      else if (mode == 2) v += bias[col];
      C[(size_t)row * ldc + col] = v;
    }
  }
}

// ---------------------------------------------------------------------------
// Batched 128x128x128 bf16 WMMA GEMM, LDS-staged via CDNA5 async copies.
// One 512-thread block (16 waves) per batch. Both 32KB tiles are staged with
// GLOBAL_LOAD_ASYNC_TO_LDS_B128 (ASYNCcnt) then consumed with ds_load
// fragments: 8 waves share the W tile, 2 waves share each A tile row-set.
// accumulate=0: store, accumulate=1: C += A*W^T
// ---------------------------------------------------------------------------
__global__ void __launch_bounds__(512)
gemm128_batched(const u16* __restrict__ A, const u16* __restrict__ W,
                float* __restrict__ C, int accumulate) {
  __shared__ __align__(16) u16 sA[16384];     // 32 KB
  __shared__ __align__(16) u16 sW[16384];     // 32 KB
  int batch = blockIdx.x;
  int tid = threadIdx.x;
  const u16* Ab = A + (size_t)batch * 16384;
  const u16* Wb = W + (size_t)batch * 16384;

  // async global -> LDS staging: 4 x b128 per buffer per thread
  #pragma unroll
  for (int j = 0; j < 4; ++j) {
    int e = (tid + j * 512) * 8;              // element index (8 u16 = 16 B)
    unsigned la = (unsigned)(uintptr_t)(void*)&sA[e];   // LDS byte offset
    unsigned lw = (unsigned)(uintptr_t)(void*)&sW[e];
    unsigned long long ga = (unsigned long long)(uintptr_t)(Ab + e);
    unsigned long long gw = (unsigned long long)(uintptr_t)(Wb + e);
    asm volatile("global_load_async_to_lds_b128 %0, %1, off"
                 :: "v"(la), "v"(ga) : "memory");
    asm volatile("global_load_async_to_lds_b128 %0, %1, off"
                 :: "v"(lw), "v"(gw) : "memory");
  }
  asm volatile("s_wait_asynccnt 0" ::: "memory");
  __syncthreads();

  int lane = tid & 31;
  int wid = tid >> 5;                 // 0..15
  int mt = wid & 7, ng = wid >> 3;
  int r0 = mt << 4, n0 = ng << 6;
  int mrow = lane & 15, hl = lane >> 4;
  float* Cb = C + (size_t)batch * 16384;

  v8f acc[4];
  #pragma unroll
  for (int t = 0; t < 4; ++t) acc[t] = (v8f)(0.0f);

  const u16* arow = &sA[(r0 + mrow) * 128];
  #pragma unroll
  for (int k0 = 0; k0 < 128; k0 += 32) {
    Frag16 af;
    af.h[0] = *(const u16x8*)(arow + k0 + hl * 8);
    af.h[1] = *(const u16x8*)(arow + k0 + 16 + hl * 8);
    #pragma unroll
    for (int t = 0; t < 4; ++t) {
      const u16* bp = &sW[(n0 + t * 16 + mrow) * 128 + k0 + hl * 8];
      Frag16 bf;
      bf.h[0] = *(const u16x8*)(bp);
      bf.h[1] = *(const u16x8*)(bp + 16);
      acc[t] = __builtin_amdgcn_wmma_f32_16x16x32_bf16(
          false, af.v, false, bf.v, (short)0, acc[t], false, false);
    }
  }
  #pragma unroll
  for (int t = 0; t < 4; ++t) {
    int col = n0 + t * 16 + mrow;
    #pragma unroll
    for (int i = 0; i < 8; ++i) {
      int row = r0 + i + (hl << 3);
      float v = acc[t][i];
      if (accumulate) v += Cb[row * 128 + col];
      Cb[row * 128 + col] = v;
    }
  }
}

// ---------------------------------------------------------------------------
// Depthwise causal conv (width 4) + bias + SiLU on xBC channels
// ---------------------------------------------------------------------------
__global__ void conv_silu_kernel(const float* __restrict__ zx,
                                 const float* __restrict__ cw,
                                 const float* __restrict__ cb,
                                 float* __restrict__ xBCc) {
  size_t idx = (size_t)blockIdx.x * blockDim.x + threadIdx.x;
  size_t total = (size_t)BSZ * LSEQ * CONVD;
  if (idx >= total) return;
  int c = (int)(idx % CONVD);
  int l = (int)((idx / CONVD) % LSEQ);
  int b = (int)(idx / ((size_t)CONVD * LSEQ));
  float acc = cb[c];
  #pragma unroll
  for (int j = 0; j < 4; ++j) {
    int ls = l - 3 + j;
    if (ls >= 0)
      acc += zx[((size_t)(b * LSEQ + ls)) * DIPP + DI + c] * cw[c * 4 + j];
  }
  xBCc[idx] = acc / (1.f + __expf(-acc));   // SiLU
}

// ---------------------------------------------------------------------------
// dt = softplus(raw + dt_bias)
// ---------------------------------------------------------------------------
__global__ void dt_kernel(const float* __restrict__ zx,
                          const float* __restrict__ dt_bias,
                          float* __restrict__ dtb) {
  int idx = blockIdx.x * blockDim.x + threadIdx.x;
  if (idx >= TOK * NH) return;
  int h = idx & 7;
  int tok = idx >> 3;
  float raw = zx[(size_t)tok * DIPP + (DI + CONVD) + h] + dt_bias[h];
  dtb[idx] = raw > 20.f ? raw : log1pf(__expf(raw));
}

// ---------------------------------------------------------------------------
// Chunk prep: per (b,h,chunk) compute inclusive cumsum of log(dA) over the
// 128-token chunk and emit bf16 operand tiles:
//   Cbf[t,n]=C, Bbf[t,n]=B, Cexp[t,n]=C*exp(cum_t),
//   Bwt[n,t]=B*dt_t*exp(cum_last-cum_t), Xt[p,t]=x_t[p]
// ---------------------------------------------------------------------------
__global__ void __launch_bounds__(128)
chunk_prep_kernel(const float* __restrict__ xBCc, const float* __restrict__ dtb,
                  const float* __restrict__ A_log,
                  u16* __restrict__ Cbf, u16* __restrict__ Bbf,
                  u16* __restrict__ Cexp, u16* __restrict__ Bwt,
                  u16* __restrict__ Xt, float* __restrict__ cum) {
  int bhc = blockIdx.x;
  int t = threadIdx.x;
  int bh = bhc >> 4, c = bhc & 15;
  int b = bh >> 3, h = bh & 7;
  int tok = b * LSEQ + c * QC + t;
  float dt = dtb[tok * NH + h];
  float la = dt * (-__expf(A_log[h]));
  __shared__ float sc[128];
  sc[t] = la; __syncthreads();
  for (int ofs = 1; ofs < 128; ofs <<= 1) {   // Hillis-Steele inclusive scan
    float v = (t >= ofs) ? sc[t - ofs] : 0.f;
    __syncthreads();
    sc[t] += v;
    __syncthreads();
  }
  float cumt = sc[t], cumlast = sc[127];
  cum[bhc * QC + t] = cumt;
  float et = __expf(cumt);                    // <= 1 (cum is non-positive)
  float w  = dt * __expf(cumlast - cumt);     // <= dt (safe)
  size_t base = (size_t)bhc * 16384;
  size_t xrow = (size_t)tok * CONVD;
  for (int n = 0; n < 128; ++n) {
    float Bv = xBCc[xrow + DI + n];
    float Cv = xBCc[xrow + DI + DSTATE + n];
    float xv = xBCc[xrow + h * HD + n];
    Cbf [base + t * 128 + n] = f2bf(Cv);
    Bbf [base + t * 128 + n] = f2bf(Bv);
    Cexp[base + t * 128 + n] = f2bf(Cv * et);
    Bwt [base + n * 128 + t] = f2bf(Bv * w);
    Xt  [base + n * 128 + t] = f2bf(xv);
  }
}

// ---------------------------------------------------------------------------
// Causal mask + decay scale: G[t,s] = CBT[t,s]*exp(cum_t-cum_s)*dt_s (s<=t)
// ---------------------------------------------------------------------------
__global__ void mask_scale_kernel(const float* __restrict__ CBT,
                                  const float* __restrict__ cum,
                                  const float* __restrict__ dtb,
                                  u16* __restrict__ G) {
  size_t idx = (size_t)blockIdx.x * blockDim.x + threadIdx.x;
  if (idx >= (size_t)NBATCH * 16384) return;
  int bhc = (int)(idx >> 14);
  int r = (int)(idx & 16383);
  int t = r >> 7, s = r & 127;
  float g = 0.f;
  if (s <= t) {
    int bh = bhc >> 4, c = bhc & 15;
    int b = bh >> 3, h = bh & 7;
    int toks = b * LSEQ + c * QC + s;
    g = CBT[idx] * __expf(cum[bhc * QC + t] - cum[bhc * QC + s]) * dtb[toks * NH + h];
  }
  G[idx] = f2bf(g);
}

// ---------------------------------------------------------------------------
// Parallel inter-chunk recurrence: each thread owns one (b,h,p,n) state
// element and walks the 16 chunks: emit S_prev (bf16) then
// S = exp(cum_last[c]) * S + S_chunk[c].
// ---------------------------------------------------------------------------
__global__ void state_serial_kernel(const float* __restrict__ Schunk,
                                    const float* __restrict__ cum,
                                    u16* __restrict__ Sprevbf) {
  int idx = blockIdx.x * blockDim.x + threadIdx.x;
  if (idx >= BSZ * NH * HD * DSTATE) return;
  int bh = idx >> 14;                 // 0..15
  int pn = idx & 16383;
  float S = 0.f;
  #pragma unroll 4
  for (int c = 0; c < NCH; ++c) {
    int bhc = bh * NCH + c;
    Sprevbf[(size_t)bhc * 16384 + pn] = f2bf(S);
    float lam = __expf(cum[bhc * QC + (QC - 1)]);
    S = lam * S + Schunk[(size_t)bhc * 16384 + pn];
  }
}

// ---------------------------------------------------------------------------
// y = (Yw + D_skip*x) * silu(z); RMS-normalize over D_INNER; * gnorm_w -> bf16
// Yw layout: [bhc][t][p]
// ---------------------------------------------------------------------------
__global__ void __launch_bounds__(256)
gate_norm_kernel(const float* __restrict__ zx, const float* __restrict__ Yw,
                 const float* __restrict__ xBCc, const float* __restrict__ Dskip,
                 const float* __restrict__ gw, u16* __restrict__ out) {
  int tok = blockIdx.x, tid = threadIdx.x;
  int b = tok >> 11;
  int tl = tok & 2047;
  int c128 = tl >> 7, t = tl & 127;
  __shared__ float red[256];
  float yv[4]; float ss = 0.f;
  #pragma unroll
  for (int j = 0; j < 4; ++j) {
    int cc = tid + j * 256;
    int h = cc >> 7, p = cc & 127;
    int bhc = ((b * NH + h) << 4) + c128;
    float y = Yw[(size_t)bhc * 16384 + t * 128 + p]
            + Dskip[h] * xBCc[(size_t)tok * CONVD + cc];
    float z = zx[(size_t)tok * DIPP + cc];
    y *= z / (1.f + __expf(-z));
    yv[j] = y; ss += y * y;
  }
  red[tid] = ss; __syncthreads();
  #pragma unroll
  for (int s = 128; s > 0; s >>= 1) { if (tid < s) red[tid] += red[tid + s]; __syncthreads(); }
  float r = rsqrtf(red[0] * (1.f / DI) + 1e-5f);
  #pragma unroll
  for (int j = 0; j < 4; ++j) {
    int cc = tid + j * 256;
    out[(size_t)tok * DI + cc] = f2bf(yv[j] * r * gw[cc]);
  }
}

// ---------------------------------------------------------------------------
extern "C" void kernel_launch(void* const* d_in, const int* in_sizes, int n_in,
                              void* d_out, int out_size, void* d_ws, size_t ws_size,
                              hipStream_t stream) {
  (void)in_sizes; (void)n_in; (void)out_size; (void)ws_size;
  const float* hidden  = (const float*)d_in[0];
  const float* w_in    = (const float*)d_in[1];
  const float* conv_w  = (const float*)d_in[2];
  const float* conv_b  = (const float*)d_in[3];
  const float* dt_bias = (const float*)d_in[4];
  const float* A_log   = (const float*)d_in[5];
  const float* D_skip  = (const float*)d_in[6];
  const float* gnorm_w = (const float*)d_in[7];
  const float* w_out   = (const float*)d_in[8];
  const float* ln1_w   = (const float*)d_in[9];
  const float* ln1_b   = (const float*)d_in[10];
  const float* ln2_w   = (const float*)d_in[11];
  const float* ln2_b   = (const float*)d_in[12];
  const float* mlp_w   = (const float*)d_in[13];
  const float* mlp_b   = (const float*)d_in[14];
  float* out = (float*)d_out;

  char* ws = (char*)d_ws;
  size_t off = 0;
  auto take = [&](size_t bytes) -> char* {
    char* p = ws + off;
    off = (off + bytes + 255) & ~(size_t)255;
    return p;
  };
  const size_t TILE = (size_t)NBATCH * 16384;     // one 128x128 tile per batch
  u16*   winbf  = (u16*)  take((size_t)DIP * DM * 2);
  u16*   woutbf = (u16*)  take((size_t)DM * DI * 2);
  u16*   mlpbf  = (u16*)  take((size_t)DM * DM * 2);
  u16*   ln1bf  = (u16*)  take((size_t)TOK * DM * 2);
  float* zx     = (float*)take((size_t)TOK * DIPP * 4);
  float* xBCc   = (float*)take((size_t)TOK * CONVD * 4);
  float* dtb    = (float*)take((size_t)TOK * NH * 4);
  u16*   Cbf    = (u16*)  take(TILE * 2);   // later aliased as G
  u16*   Bbf    = (u16*)  take(TILE * 2);   // later aliased as S_prev(bf16)
  u16*   Cexp   = (u16*)  take(TILE * 2);
  u16*   Bwt    = (u16*)  take(TILE * 2);
  u16*   Xt     = (u16*)  take(TILE * 2);
  float* cum    = (float*)take((size_t)NBATCH * QC * 4);
  float* CBT    = (float*)take(TILE * 4);   // later aliased as S_chunk
  float* Yw     = (float*)take(TILE * 4);
  u16*   ynbf   = (u16*)  take((size_t)TOK * DI * 2);
  float* res2   = (float*)take((size_t)TOK * DM * 4);
  u16*   ln2bf  = (u16*)  take((size_t)TOK * DM * 2);
  u16*   G      = Cbf;                      // alias (Cbf dead after CBT GEMM)
  u16*   Sprevbf= Bbf;                      // alias (Bbf dead after CBT GEMM)
  float* Schunk = CBT;                      // alias (CBT dead after mask)

  // weight conversions to bf16
  { int n = DIP * DM; f32_to_bf16_kernel<<<(n + 255) / 256, 256, 0, stream>>>(w_in, winbf, n); }
  { int n = DM * DI;  f32_to_bf16_kernel<<<(n + 255) / 256, 256, 0, stream>>>(w_out, woutbf, n); }
  { int n = DM * DM;  f32_to_bf16_kernel<<<(n + 255) / 256, 256, 0, stream>>>(mlp_w, mlpbf, n); }

  // ln1 -> bf16
  ln_bf16_kernel<<<TOK, 128, 0, stream>>>(hidden, ln1_w, ln1_b, ln1bf);

  // in_proj GEMM: [4096,128] x [2312,128]^T -> zx [4096,2320(pad)]
  { int mt = TOK / 16, ngr = (DIP + 63) / 64;
    int blocks = (mt * ngr + 3) / 4;
    gemm_bf16_wmma<<<blocks, 128, 0, stream>>>(ln1bf, winbf, zx, nullptr, nullptr,
                                               TOK, DIP, DM, DIPP, 0, ngr); }

  // conv + SiLU, dt
  { size_t n = (size_t)BSZ * LSEQ * CONVD;
    conv_silu_kernel<<<(int)((n + 255) / 256), 256, 0, stream>>>(zx, conv_w, conv_b, xBCc); }
  dt_kernel<<<(TOK * NH + 255) / 256, 256, 0, stream>>>(zx, dt_bias, dtb);

  // ---- chunked SSD scan (all-WMMA) ----
  chunk_prep_kernel<<<NBATCH, 128, 0, stream>>>(xBCc, dtb, A_log,
                                                Cbf, Bbf, Cexp, Bwt, Xt, cum);
  // CBT = C * B^T   [per batch 128x128x128]
  gemm128_batched<<<NBATCH, 512, 0, stream>>>(Cbf, Bbf, CBT, 0);
  // G = causal-mask + decay scale (bf16), overwrites Cbf storage
  { size_t n = TILE;
    mask_scale_kernel<<<(int)((n + 255) / 256), 256, 0, stream>>>(CBT, cum, dtb, G); }
  // Y_intra = G * X
  gemm128_batched<<<NBATCH, 512, 0, stream>>>(G, Xt, Yw, 0);
  // S_chunk = X^T * (B*w)   (overwrites CBT storage)
  gemm128_batched<<<NBATCH, 512, 0, stream>>>(Xt, Bwt, Schunk, 0);
  // inter-chunk recurrence (parallel over 262144 state elements)
  { int n = BSZ * NH * HD * DSTATE;
    state_serial_kernel<<<(n + 255) / 256, 256, 0, stream>>>(Schunk, cum, Sprevbf); }
  // Y += (C*exp(cum)) * S_prev^T
  gemm128_batched<<<NBATCH, 512, 0, stream>>>(Cexp, Sprevbf, Yw, 1);

  // gate + RMSNorm -> bf16
  gate_norm_kernel<<<TOK, 256, 0, stream>>>(zx, Yw, xBCc, D_skip, gnorm_w, ynbf);

  // out_proj GEMM + residual: [4096,1024] x [128,1024]^T + hidden -> res2
  { int mt = TOK / 16, ngr = 2;
    int blocks = (mt * ngr + 3) / 4;
    gemm_bf16_wmma<<<blocks, 128, 0, stream>>>(ynbf, woutbf, res2, hidden, nullptr,
                                               TOK, DM, DI, DM, 1, ngr); }

  // ln2 -> bf16
  ln_bf16_kernel<<<TOK, 128, 0, stream>>>(res2, ln2_w, ln2_b, ln2bf);

  // MLP GEMM + bias -> d_out
  { int mt = TOK / 16, ngr = 2;
    int blocks = (mt * ngr + 3) / 4;
    gemm_bf16_wmma<<<blocks, 128, 0, stream>>>(ln2bf, mlpbf, out, nullptr, mlp_b,
                                               TOK, DM, DM, DM, 2, ngr); }
}
